// Attention_63282048139801
// MI455X (gfx1250) — matrix operand order repaired and checked by reading, not compiled, not run on hardware
//
#include <hip/hip_runtime.h>
#include <math.h>

// ---------------------------------------------------------------------------
// Causal single-head attention, B=4, S=2048, D=1024, f32 in/out.
// 4-stage WMMA pipeline (f16 operands, f32 accumulation):
//   1) qkv_kernel    : Q/K = X*W^T (f16), V stored transposed [d][s] (f16)
//                      f32 source: register-staged cvt, double-buffered LDS
//   2) scores_kernel : S = (Q K^T)/sqrt(D) (f16); async global->LDS copies,
//                      double-buffered; upper-triangular tiles skipped
//   3) softmax_kernel: in-place causal softmax rows of S -> P, masked tail = 0
//   4) out_kernel    : O = P * V (f32 out); async copies; K-loop truncated at
//                      the causal bound
// Workspace: 80 MB (Q 16M | K 16M | Vt 16M | S/P 32M).
// ---------------------------------------------------------------------------

typedef __attribute__((ext_vector_type(16))) _Float16 v16h;
typedef __attribute__((ext_vector_type(8)))  _Float16 v8h;
typedef __attribute__((ext_vector_type(8)))  float    v8f;

#define D_MODEL 1024
#define SEQ     2048
#define NBATCH  4

#define BM 128
#define BN 128
#define BK 32
#define LDSP   40    // halves per LDS row: BK + 8 pad (rows 80B -> 16B aligned)
#define TILE_H (BM * LDSP)   // halves per tile buffer (5120)
#define TPITCH 136   // V-transpose staging pitch (halves)

// ------------------------- CDNA5 async copy helpers -------------------------
// global_load_async_to_lds_b128: per-lane 16B DMA global -> LDS, ASYNCcnt.
__device__ __forceinline__ void async_ld_b128(void* lds, const void* gsrc) {
    unsigned           loff = (unsigned)(unsigned long long)lds;  // LDS offset = flat[31:0]
    unsigned long long g    = (unsigned long long)gsrc;
    asm volatile("global_load_async_to_lds_b128 %0, %1, off"
                 :: "v"(loff), "v"(g) : "memory");
}
__device__ __forceinline__ void wait_async0() {
    asm volatile("s_wait_asynccnt 0" ::: "memory");
}

__device__ __forceinline__ v16h cat16(v8h lo, v8h hi) {
    return __builtin_shufflevector(lo, hi, 0,1,2,3,4,5,6,7,8,9,10,11,12,13,14,15);
}

// A-fragment 16x32 (MxK) f16, ISA layout:
// lane L (kb=L>>4, m=L&15): elems 0..7 = K[8kb..8kb+7], elems 8..15 = K[16+8kb..+7]
__device__ __forceinline__ v16h load_afrag(const _Float16* lds, int rowbase, int lane) {
    int row = rowbase + (lane & 15);
    int kb  = lane >> 4;
    const _Float16* p = lds + row * LDSP;
    v8h lo = *(const v8h*)(p + kb * 8);
    v8h hi = *(const v8h*)(p + 16 + kb * 8);
    return cat16(lo, hi);
}

// B-fragment 32x16 (KxN) f16 from tile stored [N][K]:
// lane L (kb=L>>4, n=L&15): elems 0..7 = K[8kb..8kb+7], elems 8..15 = K[16+8kb..+7] of column n
__device__ __forceinline__ v16h load_bfrag(const _Float16* lds, int rowbase, int lane) {
    int row = rowbase + (lane & 15);
    int kb  = lane >> 4;
    const _Float16* p = lds + row * LDSP;
    v8h lo = *(const v8h*)(p + kb * 8);
    v8h hi = *(const v8h*)(p + 16 + kb * 8);
    return cat16(lo, hi);
}

// Issue async copies for a 128x32 f16 tile (each of 256 threads moves 32B).
__device__ __forceinline__ void async_tile_f16(_Float16* lds, const _Float16* src,
                                               long long row0, int col0, int pitch, int tid) {
    int tr = tid >> 1, tc = (tid & 1) * 16;
    const _Float16* g = src + (row0 + tr) * (long long)pitch + (col0 + tc);
    _Float16*       l = lds + tr * LDSP + tc;
    async_ld_b128(l,     g);
    async_ld_b128(l + 8, g + 8);
}

// f32 source path: stage 16 floats in registers, convert+commit to LDS later.
struct RegsF32 { float4 f[4]; };
__device__ __forceinline__ RegsF32 load_regs_f32(const float* src, long long row0,
                                                 int col0, int pitch, int tid) {
    int tr = tid >> 1, tc = (tid & 1) * 16;
    const float4* p = (const float4*)(src + (row0 + tr) * (long long)pitch + (col0 + tc));
    RegsF32 r;
    r.f[0] = p[0]; r.f[1] = p[1]; r.f[2] = p[2]; r.f[3] = p[3];
    if (col0 + BK < pitch)
        __builtin_prefetch(p + 8, 0, 0);   // next chunk (global_prefetch_b8)
    return r;
}
__device__ __forceinline__ void commit_f32(_Float16* lds, const RegsF32& r, int tid) {
    int tr = tid >> 1, tc = (tid & 1) * 16;
    v8h h0, h1;
    h0[0]=(_Float16)r.f[0].x; h0[1]=(_Float16)r.f[0].y; h0[2]=(_Float16)r.f[0].z; h0[3]=(_Float16)r.f[0].w;
    h0[4]=(_Float16)r.f[1].x; h0[5]=(_Float16)r.f[1].y; h0[6]=(_Float16)r.f[1].z; h0[7]=(_Float16)r.f[1].w;
    h1[0]=(_Float16)r.f[2].x; h1[1]=(_Float16)r.f[2].y; h1[2]=(_Float16)r.f[2].z; h1[3]=(_Float16)r.f[2].w;
    h1[4]=(_Float16)r.f[3].x; h1[5]=(_Float16)r.f[3].y; h1[6]=(_Float16)r.f[3].z; h1[7]=(_Float16)r.f[3].w;
    _Float16* q = lds + tr * LDSP + tc;
    *(v8h*)q       = h0;
    *(v8h*)(q + 8) = h1;
}

// One K-chunk of WMMA work for a wave owning a 64x32 sub-tile (4x2 accs).
__device__ __forceinline__ void wave_mma(const _Float16* As, const _Float16* Bs,
                                         int wm, int wn, int lane, v8f acc[4][2]) {
    v16h a[4], bfr[2];
#pragma unroll
    for (int mi = 0; mi < 4; ++mi) a[mi] = load_afrag(As, wm + mi * 16, lane);
#pragma unroll
    for (int ni = 0; ni < 2; ++ni) bfr[ni] = load_bfrag(Bs, wn + ni * 16, lane);
#pragma unroll
    for (int mi = 0; mi < 4; ++mi)
#pragma unroll
        for (int ni = 0; ni < 2; ++ni)
            acc[mi][ni] = __builtin_amdgcn_wmma_f32_16x16x32_f16(
                false, a[mi], false, bfr[ni], (short)0, acc[mi][ni], false, false);
}

// ---------------- Stage 1: QKV projections (+ V transpose) ------------------
__global__ void __launch_bounds__(256)
qkv_kernel(const float* __restrict__ x,
           const float* __restrict__ wq, const float* __restrict__ wk, const float* __restrict__ wv,
           _Float16* __restrict__ Qb, _Float16* __restrict__ Kb, _Float16* __restrict__ Vt) {
    // 40KB arena: double-buffered A/B tiles; overlaid by V-transpose staging.
    __shared__ __align__(16) _Float16 smem[4 * TILE_H];
    _Float16* As0 = smem;                 // + cur*TILE_H selects buffer
    _Float16* Bs0 = smem + 2 * TILE_H;
    _Float16* Ts  = smem;   // 128*TPITCH = 17408 halves <= 4*TILE_H (dead after loop)

    const int tid  = threadIdx.x;
    const int lane = tid & 31;
    const int wave = tid >> 5;
    const int wm   = (wave >> 2) * 64;
    const int wn   = (wave & 3)  * 32;
    const long long m0 = (long long)blockIdx.x * BM;   // rows of flattened [B*S, D]
    const int       n0 = blockIdx.y * BN;              // output feature
    const int z = blockIdx.z;
    const float* w = (z == 0) ? wq : (z == 1) ? wk : wv;

    v8f acc[4][2] = {};
    RegsF32 ra = load_regs_f32(x, m0,            0, D_MODEL, tid);
    RegsF32 rb = load_regs_f32(w, (long long)n0, 0, D_MODEL, tid);
    commit_f32(As0, ra, tid);
    commit_f32(Bs0, rb, tid);
    __syncthreads();

    const int NC = D_MODEL / BK;
    for (int c = 0; c < NC; ++c) {
        const int cur = c & 1;
        const int nxt = cur ^ 1;
        if (c + 1 < NC) {   // issue next chunk's global loads before compute
            ra = load_regs_f32(x, m0,            (c + 1) * BK, D_MODEL, tid);
            rb = load_regs_f32(w, (long long)n0, (c + 1) * BK, D_MODEL, tid);
        }
        wave_mma(As0 + cur * TILE_H, Bs0 + cur * TILE_H, wm, wn, lane, acc);
        if (c + 1 < NC) {
            commit_f32(As0 + nxt * TILE_H, ra, tid);
            commit_f32(Bs0 + nxt * TILE_H, rb, tid);
        }
        __syncthreads();
    }

    const int kb   = lane >> 4;
    const int ncol = lane & 15;
    if (z < 2) {
        _Float16* out = (z == 0) ? Qb : Kb;
#pragma unroll
        for (int mi = 0; mi < 4; ++mi)
#pragma unroll
            for (int ni = 0; ni < 2; ++ni)
#pragma unroll
                for (int r = 0; r < 8; ++r) {
                    long long row = m0 + wm + mi * 16 + r + 8 * kb;
                    int       col = n0 + wn + ni * 16 + ncol;
                    out[row * D_MODEL + col] = (_Float16)acc[mi][ni][r];
                }
    } else {
        // stage accumulators in LDS (overlaid arena), write V transposed: Vt[b][d][s]
#pragma unroll
        for (int mi = 0; mi < 4; ++mi)
#pragma unroll
            for (int ni = 0; ni < 2; ++ni)
#pragma unroll
                for (int r = 0; r < 8; ++r) {
                    int ml = wm + mi * 16 + r + 8 * kb;
                    int nl = wn + ni * 16 + ncol;
                    Ts[ml * TPITCH + nl] = (_Float16)acc[mi][ni][r];
                }
        __syncthreads();
        const int bb    = (int)(m0 >> 11);
        const int sbase = (int)(m0 & (SEQ - 1));
#pragma unroll 4
        for (int i = 0; i < 32; ++i) {
            int idx = tid + 256 * i;          // 0..8191 -> pairs of seq positions
            int nl  = idx >> 6;               // local feature 0..127
            int m2  = (idx & 63) * 2;         // local seq (even)
            unsigned short u0 = *(const unsigned short*)&Ts[m2 * TPITCH + nl];
            unsigned short u1 = *(const unsigned short*)&Ts[(m2 + 1) * TPITCH + nl];
            int d = n0 + nl, s = sbase + m2;
            unsigned* dst = (unsigned*)(Vt + ((long long)bb * D_MODEL + d) * SEQ + s);
            *dst = (unsigned)u0 | ((unsigned)u1 << 16);
        }
    }
}

// ------ Generic f16 GEMM mainloop: async copies + double-buffered LDS -------
__device__ __forceinline__ void gemm_f16_mainloop(const _Float16* A, const _Float16* Bp,
                                                  long long m0, int n0,
                                                  int pitchA, int pitchB, int kdim,
                                                  _Float16* As0, _Float16* Bs0,
                                                  int wm, int wn, int lane, int tid,
                                                  v8f acc[4][2]) {
    async_tile_f16(As0, A,  m0,            0, pitchA, tid);
    async_tile_f16(Bs0, Bp, (long long)n0, 0, pitchB, tid);
    wait_async0();
    __syncthreads();

    const int NC = kdim / BK;
    for (int c = 0; c < NC; ++c) {
        const int cur = c & 1;
        const int nxt = cur ^ 1;
        if (c + 1 < NC) {   // DMA next chunk into the other buffer
            async_tile_f16(As0 + nxt * TILE_H, A,  m0,            (c + 1) * BK, pitchA, tid);
            async_tile_f16(Bs0 + nxt * TILE_H, Bp, (long long)n0, (c + 1) * BK, pitchB, tid);
        }
        wave_mma(As0 + cur * TILE_H, Bs0 + cur * TILE_H, wm, wn, lane, acc);
        wait_async0();
        __syncthreads();
    }
}

// ---------------- Stage 2: scores = Q K^T * scale (causal tiles only) -------
__global__ void __launch_bounds__(256)
scores_kernel(const _Float16* __restrict__ Q, const _Float16* __restrict__ K,
              _Float16* __restrict__ S) {
    const int kt = blockIdx.x, qt = blockIdx.y, b = blockIdx.z;
    if (kt > qt) return;  // entirely above causal diagonal

    __shared__ __align__(16) _Float16 smem[4 * TILE_H];
    _Float16* As0 = smem;
    _Float16* Bs0 = smem + 2 * TILE_H;

    const int tid = threadIdx.x, lane = tid & 31, wave = tid >> 5;
    const int wm = (wave >> 2) * 64, wn = (wave & 3) * 32;
    const _Float16* A  = Q + (long long)b * SEQ * D_MODEL;
    const _Float16* Bp = K + (long long)b * SEQ * D_MODEL;
    _Float16*       So = S + (long long)b * SEQ * SEQ;
    const long long m0 = (long long)qt * BM;
    const int       n0 = kt * BN;

    v8f acc[4][2] = {};
    gemm_f16_mainloop(A, Bp, m0, n0, D_MODEL, D_MODEL, D_MODEL,
                      As0, Bs0, wm, wn, lane, tid, acc);

    const float scale = 0.03125f;  // 1/sqrt(1024)
    const int kb = lane >> 4, ncol = lane & 15;
#pragma unroll
    for (int mi = 0; mi < 4; ++mi)
#pragma unroll
        for (int ni = 0; ni < 2; ++ni)
#pragma unroll
            for (int r = 0; r < 8; ++r) {
                int row = (int)m0 + wm + mi * 16 + r + 8 * kb;
                int col = n0 + wn + ni * 16 + ncol;
                So[(long long)row * SEQ + col] = (_Float16)(acc[mi][ni][r] * scale);
            }
}

// ---------------- Stage 3: causal softmax (in place, f16) -------------------
__global__ void __launch_bounds__(256)
softmax_kernel(_Float16* __restrict__ S) {
    const long long rowid = blockIdx.x;              // 0..8191
    const int b = (int)(rowid >> 11);
    const int q = (int)(rowid & (SEQ - 1));
    _Float16* row = S + ((long long)b * SEQ + q) * SEQ;
    const int L = q + 1;
    const int t = threadIdx.x;
    __shared__ float red[256];

    float mx = -1e30f;
    for (int i = t; i < L; i += 256) mx = fmaxf(mx, (float)row[i]);
    red[t] = mx; __syncthreads();
    for (int s = 128; s > 0; s >>= 1) { if (t < s) red[t] = fmaxf(red[t], red[t + s]); __syncthreads(); }
    mx = red[0]; __syncthreads();

    float sum = 0.f;
    for (int i = t; i < L; i += 256) sum += __expf((float)row[i] - mx);
    red[t] = sum; __syncthreads();
    for (int s = 128; s > 0; s >>= 1) { if (t < s) red[t] += red[t + s]; __syncthreads(); }
    const float inv = 1.0f / red[0];

    for (int i = t; i < L; i += 256) row[i] = (_Float16)(__expf((float)row[i] - mx) * inv);
    for (int i = L + t; i < SEQ; i += 256) row[i] = (_Float16)0.f;  // masked tail
}

// ---------------- Stage 4: O = P * V (causal-truncated K loop) --------------
__global__ void __launch_bounds__(256)
out_kernel(const _Float16* __restrict__ P, const _Float16* __restrict__ Vt,
           float* __restrict__ O) {
    const int nt = blockIdx.x, qt = blockIdx.y, b = blockIdx.z;

    __shared__ __align__(16) _Float16 smem[4 * TILE_H];
    _Float16* As0 = smem;
    _Float16* Bs0 = smem + 2 * TILE_H;

    const int tid = threadIdx.x, lane = tid & 31, wave = tid >> 5;
    const int wm = (wave >> 2) * 64, wn = (wave & 3) * 32;
    const _Float16* A  = P  + (long long)b * SEQ * SEQ;       // [q][k], pitch SEQ
    const _Float16* Bp = Vt + (long long)b * D_MODEL * SEQ;   // [d][k], pitch SEQ
    const long long m0 = (long long)qt * BM;
    const int       n0 = nt * BN;
    const int kmax = (qt + 1) * BM;   // columns beyond the causal bound are zero

    v8f acc[4][2] = {};
    gemm_f16_mainloop(A, Bp, m0, n0, SEQ, SEQ, kmax,
                      As0, Bs0, wm, wn, lane, tid, acc);

    const int kb = lane >> 4, ncol = lane & 15;
#pragma unroll
    for (int mi = 0; mi < 4; ++mi)
#pragma unroll
        for (int ni = 0; ni < 2; ++ni)
#pragma unroll
            for (int r = 0; r < 8; ++r) {
                int row = (int)m0 + wm + mi * 16 + r + 8 * kb;
                int col = n0 + wn + ni * 16 + ncol;
                O[((long long)b * SEQ + row) * D_MODEL + col] = acc[mi][ni][r];
            }
}

// ---------------------------------------------------------------------------
extern "C" void kernel_launch(void* const* d_in, const int* in_sizes, int n_in,
                              void* d_out, int out_size, void* d_ws, size_t ws_size,
                              hipStream_t stream) {
    const float* x  = (const float*)d_in[0];
    const float* wq = (const float*)d_in[1];
    const float* wk = (const float*)d_in[2];
    const float* wv = (const float*)d_in[3];

    char* ws = (char*)d_ws;
    _Float16* Qb = (_Float16*)(ws);                               // 16 MB
    _Float16* Kb = (_Float16*)(ws + (size_t)16 * 1024 * 1024);    // 16 MB
    _Float16* Vt = (_Float16*)(ws + (size_t)32 * 1024 * 1024);    // 16 MB
    _Float16* Sb = (_Float16*)(ws + (size_t)48 * 1024 * 1024);    // 32 MB
    float*    O  = (float*)d_out;

    dim3 g1((NBATCH * SEQ) / BM, D_MODEL / BN, 3);
    qkv_kernel<<<g1, 256, 0, stream>>>(x, wq, wk, wv, Qb, Kb, Vt);

    dim3 g2(SEQ / BN, SEQ / BM, NBATCH);
    scores_kernel<<<g2, 256, 0, stream>>>(Qb, Kb, Sb);

    softmax_kernel<<<NBATCH * SEQ, 256, 0, stream>>>(Sb);

    dim3 g3(D_MODEL / BN, SEQ / BM, NBATCH);
    out_kernel<<<g3, 256, 0, stream>>>(Sb, Vt, O);
}
